// DeltaNetEnhancedMoE_69569880261169
// MI455X (gfx1250) — matrix operand order repaired and checked by reading, not compiled, not run on hardware
//
#include <hip/hip_runtime.h>
#include <hip/hip_bf16.h>
#include <math.h>

#define T_TOK     16384
#define C_DIM     1024
#define E_NUM     8
#define CAPACITY  2560
#define INTER_DIM 2730
#define TWO_I     5460
#define GCOLS     2752      // INTER padded to 86*32 for K=32 WMMA
#define GSTR      2752
#define CHALF     512
#define GIN       1030
#define GKPAD     1056      // 33*32

typedef __attribute__((ext_vector_type(16))) __bf16       v16bf;
typedef __attribute__((ext_vector_type(2)))  __bf16       v2bf;
typedef __attribute__((ext_vector_type(8)))  float        v8f;
typedef __attribute__((ext_vector_type(4)))  unsigned int u32x4;
typedef __attribute__((ext_vector_type(4)))  float        f32x4;

union BFrag { v16bf v; u32x4 q[2]; };

__device__ inline v8f v8f_zero() {
  v8f z = {0.f,0.f,0.f,0.f,0.f,0.f,0.f,0.f};
  return z;
}
// packed fp32x2 -> bf16x2: prefer the named builtin, else native __bf16
// conversion (backend selects v_cvt_pk_bf16_f32-class ops when available)
__device__ inline unsigned int pack2bf(float a, float b) {
#if __has_builtin(__builtin_amdgcn_cvt_pk_bf16_f32)
  union { v2bf v; unsigned int u; } cv;
  cv.v = __builtin_amdgcn_cvt_pk_bf16_f32(a, b);
  return cv.u;
#else
  union { v2bf v; unsigned int u; } cv;
  cv.v = v2bf{ (__bf16)a, (__bf16)b };
  return cv.u;
#endif
}
__device__ inline unsigned short f2bf(float f) {
  union { __bf16 h; unsigned short s; } cv;
  cv.h = (__bf16)f;
  return cv.s;
}
__device__ inline float bf2f(unsigned short s) {
  return __uint_as_float(((unsigned int)s) << 16);
}
__device__ inline v8f wmma_bf16(const BFrag& a, const BFrag& b, v8f c) {
  return __builtin_amdgcn_wmma_f32_16x16x32_bf16(false, a.v, false, b.v,
                                                 (short)0, c, false, false);
}
// A fragment (16x32 bf16, row-major LDS row `row`, K base kc)
__device__ inline BFrag load_fragA(const unsigned short* row, int kc, int hi) {
  BFrag f;
  f.q[0] = *(const u32x4*)(row + kc + hi * 8);
  f.q[1] = *(const u32x4*)(row + kc + 16 + hi * 8);
  return f;
}
// B fragment (32x16 bf16) from LDS stored as [col][32 k]
__device__ inline BFrag load_fragB(const unsigned short* colRow, int hi) {
  BFrag f;
  f.q[0] = *(const u32x4*)(colRow + hi * 8);
  f.q[1] = *(const u32x4*)(colRow + 16 + hi * 8);
  return f;
}

// ---------------------------------------------------------------- zero/init
__global__ void zero_kernel(float* __restrict__ out, long n, int* __restrict__ counts) {
  long i = (long)blockIdx.x * blockDim.x + threadIdx.x;
  long stride = (long)gridDim.x * blockDim.x;
  for (; i < n; i += stride) out[i] = 0.f;
  if (blockIdx.x == 0 && threadIdx.x < E_NUM) counts[threadIdx.x] = 0;
}

// ---------------------------------------------------- gating (stats + MLP + top2)
__global__ void __launch_bounds__(256)
gating_kernel(const float* __restrict__ x,
              const float* __restrict__ gw1,
              const float* __restrict__ gb1,
              const float* __restrict__ gw2,
              int* __restrict__ tkI, float* __restrict__ tkP) {
  extern __shared__ char smem[];
  unsigned short* xgS = (unsigned short*)smem;        // [16][GKPAD] bf16 gate_in
  unsigned short* wS  = xgS + 16 * GKPAD;             // [512][32]   bf16 gw1 chunk (transposed)
  unsigned short* hS  = wS + 512 * 32;                // [16][512]   bf16 hidden
  float* lgS = (float*)(hS + 16 * 512);               // [16][8]     logits

  const int tid  = threadIdx.x;
  const int wave = tid >> 5, lane = tid & 31;
  const int tok0 = blockIdx.x * 16;

  // load 16 token rows (32 contiguous cols / lane), compute stats per token
  for (int rep = 0; rep < 2; ++rep) {
    int tl = wave * 2 + rep;
    const float* xr = x + (size_t)(tok0 + tl) * C_DIM + lane * 32;
    float s = 0.f, ss = 0.f, mn = 3.4e38f, mx = -3.4e38f, cnt = 0.f;
#pragma unroll
    for (int oct = 0; oct < 4; ++oct) {
      f32x4 a = *(const f32x4*)(xr + oct * 8);
      f32x4 b = *(const f32x4*)(xr + oct * 8 + 4);
#pragma unroll
      for (int j = 0; j < 4; ++j) {
        float va = a[j], vb = b[j];
        s += va + vb; ss += va * va + vb * vb;
        mn = fminf(mn, fminf(va, vb)); mx = fmaxf(mx, fmaxf(va, vb));
        cnt += ((fabsf(va) < 1e-6f) ? 1.f : 0.f) + ((fabsf(vb) < 1e-6f) ? 1.f : 0.f);
      }
      u32x4 q = { pack2bf(a[0], a[1]), pack2bf(a[2], a[3]),
                  pack2bf(b[0], b[1]), pack2bf(b[2], b[3]) };
      *(u32x4*)(xgS + tl * GKPAD + lane * 32 + oct * 8) = q;
    }
#pragma unroll
    for (int off = 16; off; off >>= 1) {
      s   += __shfl_xor(s, off);
      ss  += __shfl_xor(ss, off);
      mn   = fminf(mn, __shfl_xor(mn, off));
      mx   = fmaxf(mx, __shfl_xor(mx, off));
      cnt += __shfl_xor(cnt, off);
    }
    float mean = s * (1.f / 1024.f);
    float var  = fmaxf((ss - 1024.f * mean * mean) * (1.f / 1023.f), 0.f);
    float stat = (lane == 0) ? mean
               : (lane == 1) ? sqrtf(var)
               : (lane == 2) ? mn
               : (lane == 3) ? mx
               : (lane == 4) ? sqrtf(ss)
                             : cnt * (1.f / 1024.f);
    xgS[tl * GKPAD + 1024 + lane] = (lane < 6) ? f2bf(stat) : (unsigned short)0;
  }
  __syncthreads();

  // GEMM1: h[16,512] = gate_in[16,1056pad] @ gw1; 4 n-tiles / wave
  v8f acc[4];
#pragma unroll
  for (int nt = 0; nt < 4; ++nt) acc[nt] = v8f_zero();
  const int m = lane & 15, hi = lane >> 4;

  for (int kc = 0; kc < GKPAD; kc += 32) {
    // stage gw1[kc..kc+31, 0:512] transposed, packed b128 stores, branch-free
    for (int rep = 0; rep < 4; ++rep) {
      int task  = tid + 256 * rep;       // 0..1023
      int c     = task >> 1;             // column 0..511
      int khalf = task & 1;
      const float* colp = gw1 + c;
      int kbase = kc + khalf * 16;
      unsigned int p[8];
#pragma unroll
      for (int jj = 0; jj < 8; ++jj) {
        int k0 = kbase + 2 * jj;
        float a = (k0     < GIN) ? colp[(size_t)k0 * CHALF] : 0.f;
        float b = (k0 + 1 < GIN) ? colp[(size_t)(k0 + 1) * CHALF] : 0.f;
        p[jj] = pack2bf(a, b);
      }
      u32x4 q0 = { p[0], p[1], p[2], p[3] };
      u32x4 q1 = { p[4], p[5], p[6], p[7] };
      *(u32x4*)(wS + c * 32 + khalf * 16)     = q0;
      *(u32x4*)(wS + c * 32 + khalf * 16 + 8) = q1;
    }
    __syncthreads();
    BFrag a = load_fragA(xgS + m * GKPAD, kc, hi);
#pragma unroll
    for (int nt = 0; nt < 4; ++nt) {
      int col = (wave * 4 + nt) * 16 + m;
      BFrag b = load_fragB(wS + col * 32, hi);
      acc[nt] = wmma_bf16(a, b, acc[nt]);
    }
    __syncthreads();
  }
  // bias + exact gelu -> hS (bf16)
#pragma unroll
  for (int nt = 0; nt < 4; ++nt) {
    int col = (wave * 4 + nt) * 16 + m;
    float bias = gb1[col];
#pragma unroll
    for (int vv = 0; vv < 8; ++vv) {
      int mm = vv + hi * 8;
      float hv = acc[nt][vv] + bias;
      hv = 0.5f * hv * (1.f + erff(hv * 0.70710678f));
      hS[mm * CHALF + col] = f2bf(hv);
    }
  }
  __syncthreads();

  // GEMM2 (512x8, tiny): 128 threads = 16 tokens x 8 experts
  if (tid < 128) {
    int tl = tid >> 3, e = tid & 7;
    float d = 0.f;
    for (int o8 = 0; o8 < CHALF / 8; ++o8) {
      u32x4 q = *(const u32x4*)(hS + tl * CHALF + o8 * 8);
#pragma unroll
      for (int jj = 0; jj < 4; ++jj) {
        unsigned int w2 = q[jj];
        d += bf2f((unsigned short)(w2 & 0xffffu)) * gw2[(o8 * 8 + 2 * jj) * E_NUM + e];
        d += bf2f((unsigned short)(w2 >> 16))     * gw2[(o8 * 8 + 2 * jj + 1) * E_NUM + e];
      }
    }
    lgS[tl * E_NUM + e] = d;  // TEMPERATURE == 1
  }
  __syncthreads();

  // softmax + top-2 + renorm, one thread per token
  if (tid < 16) {
    int t = tok0 + tid;
    float p[E_NUM];
    float mxl = -3.4e38f;
    for (int e = 0; e < E_NUM; ++e) mxl = fmaxf(mxl, lgS[tid * E_NUM + e]);
    float ssum = 0.f;
    for (int e = 0; e < E_NUM; ++e) { p[e] = __expf(lgS[tid * E_NUM + e] - mxl); ssum += p[e]; }
    float inv = 1.f / ssum;
    for (int e = 0; e < E_NUM; ++e) p[e] *= inv;
    int i1 = 0; float p1 = p[0];
    for (int e = 1; e < E_NUM; ++e) if (p[e] > p1) { p1 = p[e]; i1 = e; }
    int i2 = -1; float p2 = -1.f;
    for (int e = 0; e < E_NUM; ++e) if (e != i1 && p[e] > p2) { p2 = p[e]; i2 = e; }
    float den = fmaxf(p1 + p2, 1e-12f);
    tkI[2 * t] = i1; tkI[2 * t + 1] = i2;
    tkP[2 * t] = p1 / den; tkP[2 * t + 1] = p2 / den;
  }
}

// ------------------------------------------------- capacity slot assignment
__global__ void assign_kernel(const int* __restrict__ tkI, const float* __restrict__ tkP,
                              int* __restrict__ counts, int* __restrict__ pos,
                              float* __restrict__ gwA) {
  int t = blockIdx.x * blockDim.x + threadIdx.x;
  if (t >= T_TOK) return;
  for (int k = 0; k < 2; ++k) {
    int e = tkI[2 * t + k];
    float w = tkP[2 * t + k];
    int slot = atomicAdd(&counts[e], 1);
    if (slot < CAPACITY) {
      pos[e * CAPACITY + slot] = t;
      gwA[e * CAPACITY + slot] = w;
    }
  }
}

// ---------------------------------------- fused SwiGLU expert (bf16 WMMA)
__global__ void __launch_bounds__(256)
expert_kernel(const float* __restrict__ x,
              const float* __restrict__ W1,
              const float* __restrict__ W2,
              const int* __restrict__ pos,
              const float* __restrict__ gwA,
              const int* __restrict__ counts,
              float* __restrict__ out) {
  extern __shared__ char smem[];
  unsigned short* gS = (unsigned short*)smem;      // [16][GSTR]  bf16 g
  unsigned short* xS = gS + 16 * GSTR;             // [16][1024]  bf16 X tile
  unsigned short* wS = xS + 16 * 1024;             // 64 KB stage region
  float* uS   = (float*)wS;                        // reuse: 8 tiles x 256 f32
  int*   tokS = (int*)(wS + 32768);                // [16]
  float* gwS  = (float*)(tokS + 16);               // [16]

  const int tid  = threadIdx.x;
  const int wave = tid >> 5, lane = tid & 31;
  const int e  = blockIdx.x / (CAPACITY / 16);
  const int mt = blockIdx.x % (CAPACITY / 16);
  const float* W1e = W1 + (size_t)e * C_DIM * TWO_I;
  const float* W2e = W2 + (size_t)e * INTER_DIM * C_DIM;

  if (tid < 16) {
    int cnt  = counts[e]; if (cnt > CAPACITY) cnt = CAPACITY;
    int slot = mt * 16 + tid;
    int tk   = (slot < cnt) ? pos[e * CAPACITY + slot] : -1;
    tokS[tid] = tk;
    gwS[tid]  = (slot < cnt) ? gwA[e * CAPACITY + slot] : 0.f;
  }
  __syncthreads();

  {  // load X tile -> bf16 LDS, packed (invalid rows -> zeros)
    int row = tid >> 4;
    int cs  = (tid & 15) * 64;
    int tk  = tokS[row];
    const float* xr = x + (size_t)(tk < 0 ? 0 : tk) * C_DIM + cs;
    for (int oct = 0; oct < 8; ++oct) {
      f32x4 a = *(const f32x4*)(xr + oct * 8);
      f32x4 b = *(const f32x4*)(xr + oct * 8 + 4);
      if (tk < 0) { a = f32x4{0.f,0.f,0.f,0.f}; b = f32x4{0.f,0.f,0.f,0.f}; }
      u32x4 q = { pack2bf(a[0], a[1]), pack2bf(a[2], a[3]),
                  pack2bf(b[0], b[1]), pack2bf(b[2], b[3]) };
      *(u32x4*)(xS + row * C_DIM + cs + oct * 8) = q;
    }
  }
  __syncthreads();

  const int m = lane & 15, hi = lane >> 4;
  const int icL    = wave >> 2;
  const int isUp   = (wave >> 1) & 1;
  const int nHalf  = wave & 1;
  const int clBase = (isUp << 6) + (icL << 5) + (nHalf << 4);

  // ---- Phase 1: g = silu(X@W1g) * (X@W1u) -> gS[16][GSTR]
  for (int ic2 = 0; ic2 < 86; ic2 += 2) {
    v8f acc = v8f_zero();
    for (int kc = 0; kc < C_DIM; kc += 32) {
      {  // stage W1[kc..kc+31, 128 cols] transposed, packed; 1 task / thread
        int cl    = tid >> 1;          // 0..127
        int khalf = tid & 1;
        int isUpC  = cl >> 6;
        int within = cl & 63;
        int colOff = ic2 * 32 + within;
        bool valid = colOff < INTER_DIM;
        const float* colp = W1e + (size_t)(isUpC ? INTER_DIM : 0) + colOff;
        int kbase = kc + khalf * 16;
        unsigned int p[8];
#pragma unroll
        for (int jj = 0; jj < 8; ++jj) {
          float a = valid ? colp[(size_t)(kbase + 2 * jj) * TWO_I] : 0.f;
          float b = valid ? colp[(size_t)(kbase + 2 * jj + 1) * TWO_I] : 0.f;
          p[jj] = pack2bf(a, b);
        }
        u32x4 q0 = { p[0], p[1], p[2], p[3] };
        u32x4 q1 = { p[4], p[5], p[6], p[7] };
        *(u32x4*)(wS + cl * 32 + khalf * 16)     = q0;
        *(u32x4*)(wS + cl * 32 + khalf * 16 + 8) = q1;
        // prefetch next K-chunk rows of the same column slice (L2 -> near)
        if (kc + 32 < C_DIM && valid)
          __builtin_prefetch(colp + (size_t)(kbase + 32) * TWO_I, 0, 1);
      }
      __syncthreads();
      BFrag a = load_fragA(xS + m * C_DIM, kc, hi);
      BFrag b = load_fragB(wS + (clBase + m) * 32, hi);
      acc = wmma_bf16(a, b, acc);
      __syncthreads();
    }
    {  // spill U tiles to LDS (C layout: vgpr v -> M=v+hi*8, N=lane&15)
      int n = lane & 15;
#pragma unroll
      for (int vv = 0; vv < 8; ++vv) uS[wave * 256 + (vv + hi * 8) * 16 + n] = acc[vv];
    }
    __syncthreads();
    {  // g = silu(u_gate) * u_up -> gS bf16 (packed pairs along jc)
#pragma unroll
      for (int r = 0; r < 2; ++r) {
        int idx = tid * 2 + r;          // 0..511 pair index (2 cols each)
        int ic  = idx >> 8;             // 0/1
        int rem = idx & 255;
        int mm  = rem >> 4;
        int jp  = rem & 15;             // pair of cols 2*jp, 2*jp+1
        int nh0 = (2 * jp) >> 4, nn0 = (2 * jp) & 15;
        int nh1 = (2 * jp + 1) >> 4, nn1 = (2 * jp + 1) & 15;
        float u1a = uS[(ic * 4 + nh0) * 256 + mm * 16 + nn0];
        float u2a = uS[(ic * 4 + 2 + nh0) * 256 + mm * 16 + nn0];
        float u1b = uS[(ic * 4 + nh1) * 256 + mm * 16 + nn1];
        float u2b = uS[(ic * 4 + 2 + nh1) * 256 + mm * 16 + nn1];
        float ga = (u1a / (1.f + __expf(-u1a))) * u2a;
        float gb = (u1b / (1.f + __expf(-u1b))) * u2b;
        *(unsigned int*)(gS + mm * GSTR + (ic2 + ic) * 32 + 2 * jp) = pack2bf(ga, gb);
      }
    }
    __syncthreads();
  }

  // ---- Phase 2: out[16,1024] = g[16,2752] @ W2[2752pad,1024]; 8 n-tiles / wave
  v8f oacc[8];
#pragma unroll
  for (int nt = 0; nt < 8; ++nt) oacc[nt] = v8f_zero();
  for (int kc = 0; kc < GCOLS; kc += 32) {
    // stage W2[kc..kc+31, 0:1024] transposed, packed b128 stores
    for (int rep = 0; rep < 8; ++rep) {
      int task  = tid + 256 * rep;     // 0..2047
      int c     = task >> 1;           // 0..1023
      int khalf = task & 1;
      const float* colp = W2e + c;
      int kbase = kc + khalf * 16;
      unsigned int p[8];
#pragma unroll
      for (int jj = 0; jj < 8; ++jj) {
        int k0 = kbase + 2 * jj;
        float a = (k0     < INTER_DIM) ? colp[(size_t)k0 * C_DIM] : 0.f;
        float b = (k0 + 1 < INTER_DIM) ? colp[(size_t)(k0 + 1) * C_DIM] : 0.f;
        p[jj] = pack2bf(a, b);
      }
      u32x4 q0 = { p[0], p[1], p[2], p[3] };
      u32x4 q1 = { p[4], p[5], p[6], p[7] };
      *(u32x4*)(wS + c * 32 + khalf * 16)     = q0;
      *(u32x4*)(wS + c * 32 + khalf * 16 + 8) = q1;
    }
    {  // prefetch next K-chunk of W2 (rows kc+32..kc+63), one 512B line / thread
      int kg = kc + 32 + (tid >> 3);
      if (kg < INTER_DIM)
        __builtin_prefetch(W2e + (size_t)kg * C_DIM + (tid & 7) * 128, 0, 1);
    }
    __syncthreads();
    BFrag a = load_fragA(gS + m * GSTR, kc, hi);
#pragma unroll
    for (int nt = 0; nt < 8; ++nt) {
      int col = (wave * 8 + nt) * 16 + m;
      BFrag b = load_fragB(wS + col * 32, hi);
      oacc[nt] = wmma_bf16(a, b, oacc[nt]);
    }
    __syncthreads();
  }

  // ---- epilogue: gated scatter-add combine
  {
    int n = lane & 15;
#pragma unroll
    for (int nt = 0; nt < 8; ++nt) {
      int col = (wave * 8 + nt) * 16 + n;
#pragma unroll
      for (int vv = 0; vv < 8; ++vv) {
        int mm = vv + hi * 8;
        int tk = tokS[mm];
        if (tk >= 0) atomicAdd(&out[(size_t)tk * C_DIM + col], oacc[nt][vv] * gwS[mm]);
      }
    }
  }
}

// ---------------------------------------------------------------- launcher
#define GATING_SMEM (16 * GKPAD * 2 + 512 * 32 * 2 + 16 * 512 * 2 + 16 * 8 * 4)
#define EXPERT_SMEM (16 * GSTR * 2 + 16 * 1024 * 2 + 65536 + 16 * 4 + 16 * 4)

extern "C" void kernel_launch(void* const* d_in, const int* in_sizes, int n_in,
                              void* d_out, int out_size, void* d_ws, size_t ws_size,
                              hipStream_t stream) {
  (void)in_sizes; (void)n_in; (void)ws_size;
  const float* x   = (const float*)d_in[0];
  // d_in[1]=t, d_in[2]=snr_threshold : unused by the reference math
  const float* gw1 = (const float*)d_in[3];
  const float* gb1 = (const float*)d_in[4];
  const float* gw2 = (const float*)d_in[5];
  const float* W1  = (const float*)d_in[6];
  const float* W2  = (const float*)d_in[7];
  float* out = (float*)d_out;

  int*   counts = (int*)d_ws;                 // [8]
  int*   tkI    = counts + 8;                 // [T,2]
  float* tkP    = (float*)(tkI + 2 * T_TOK);  // [T,2]
  int*   pos    = (int*)(tkP + 2 * T_TOK);    // [E,CAP]
  float* gwA    = (float*)(pos + E_NUM * CAPACITY);  // [E,CAP]

  (void)hipFuncSetAttribute(reinterpret_cast<const void*>(gating_kernel),
                            hipFuncAttributeMaxDynamicSharedMemorySize, GATING_SMEM);
  (void)hipFuncSetAttribute(reinterpret_cast<const void*>(expert_kernel),
                            hipFuncAttributeMaxDynamicSharedMemorySize, EXPERT_SMEM);

  zero_kernel<<<2048, 256, 0, stream>>>(out, (long)out_size, counts);
  gating_kernel<<<T_TOK / 16, 256, GATING_SMEM, stream>>>(x, gw1, gb1, gw2, tkI, tkP);
  assign_kernel<<<T_TOK / 256, 256, 0, stream>>>(tkI, tkP, counts, pos, gwA);
  expert_kernel<<<E_NUM * (CAPACITY / 16), 256, EXPERT_SMEM, stream>>>(
      x, W1, W2, pos, gwA, counts, out);
}
